// GraphGPSLayer_11553462026953
// MI455X (gfx1250) — matrix operand order repaired and checked by reading, not compiled, not run on hardware
//
#include <hip/hip_runtime.h>
#include <math.h>

typedef __bf16 bf16;
typedef __attribute__((ext_vector_type(16))) __bf16 v16bf;
typedef __attribute__((ext_vector_type(8)))  __bf16 v8bf;
typedef __attribute__((ext_vector_type(8)))  float  v8f;

#define DMODEL 256
#define NNODE  4096
#define NEDGE  131072
#define NHEAD  8
#define DFFN   1024

__device__ __forceinline__ int lane_id() { return threadIdx.x & 31; }

__device__ __forceinline__ v16bf combine16(v8bf lo, v8bf hi) {
  v16bf r;
#pragma unroll
  for (int i = 0; i < 8; ++i) { r[i] = lo[i]; r[i + 8] = hi[i]; }
  return r;
}

// A fragment: 16x32 bf16 (M x K), base points at (row0,col0) of tile, row-major with leading dim `ld`.
// Layout: lanes 0-15 hold rows M=lane with K {0..15}; lanes 16-31 K {16..31}.
__device__ __forceinline__ v16bf load_a_frag(const bf16* base, int ld) {
  const int l = lane_id();
  const bf16* p = base + (size_t)(l & 15) * ld + ((l >> 4) * 16);
  v8bf lo = *(const v8bf*)p;          // K = klo .. klo+7
  v8bf hi = *(const v8bf*)(p + 8);    // K = klo+8 .. klo+15
  return combine16(lo, hi);
}

// B fragment: 32x16 bf16 (K x N) where element(k,n) lives at base[n*ld + k]
// (i.e. the K dimension is contiguous in memory). lane l -> n = l&15, k = (l>>4)*16 + e.
__device__ __forceinline__ v16bf load_bt_frag(const bf16* base, size_t ld) {
  const int l = lane_id();
  const bf16* p = base + (size_t)(l & 15) * ld + ((l >> 4) * 16);
  v8bf lo = *(const v8bf*)p;
  v8bf hi = *(const v8bf*)(p + 8);
  return combine16(lo, hi);
}

// B fragment from pre-swizzled weights: tile (kt,nt) stored as 32 lanes x 16 contiguous bf16.
__device__ __forceinline__ v16bf load_bw_frag(const bf16* Wsw, int NT, int kt, int nt) {
  const bf16* p = Wsw + (((size_t)kt * NT + nt) * 32 + lane_id()) * 16;
  v8bf lo = *(const v8bf*)p;
  v8bf hi = *(const v8bf*)(p + 8);
  return combine16(lo, hi);
}

__device__ __forceinline__ v8f wmma_bf16(v16bf a, v16bf b, v8f c) {
  return __builtin_amdgcn_wmma_f32_16x16x32_bf16(false, a, false, b, (short)0, c, false, false);
}

__device__ __forceinline__ float gelu_exact(float x) {
  return 0.5f * x * (1.0f + erff(x * 0.70710678118654752f));
}

// CDNA5 async memory->LDS copy (16B per lane), tracked by ASYNCcnt.
// LDS byte address = low 32 bits of generic shared pointer.
__device__ __forceinline__ void async_load_b128(const bf16* gptr, bf16* lptr) {
  unsigned lds = (unsigned)(uintptr_t)lptr;
  unsigned long long ga = (unsigned long long)(uintptr_t)gptr;
  asm volatile("global_load_async_to_lds_b128 %0, %1, off"
               :: "v"(lds), "v"(ga)
               : "memory");
}
__device__ __forceinline__ void wait_async0() {
  asm volatile("s_wait_asynccnt 0" ::: "memory");
}

// ---------------------------------------------------------------------------
// Weight swizzle: f32 [K, Nout] row-major -> bf16 B-fragment order.
// out index = ((kt*NT + nt)*32 + lane)*16 + e  with  k = kt*32+(lane>>4)*16+e, n = nt*16+(lane&15)
// ---------------------------------------------------------------------------
__global__ void swizzle_w_kernel(const float* __restrict__ W, bf16* __restrict__ out,
                                 int K, int Nout) {
  size_t tid = (size_t)blockIdx.x * blockDim.x + threadIdx.x;
  size_t tile = tid >> 9;
  int r = (int)(tid & 511);
  int lane = r >> 4, e = r & 15;
  int NT = Nout >> 4;
  int kt = (int)(tile / NT), nt = (int)(tile % NT);
  int k = kt * 32 + (lane >> 4) * 16 + e;
  int n = nt * 16 + (lane & 15);
  out[tid] = (bf16)W[(size_t)k * Nout + n];
}

// ---------------------------------------------------------------------------
// LayerNorm over D=256, one row per block (256 threads).
// ---------------------------------------------------------------------------
__global__ __launch_bounds__(256) void layernorm_kernel(
    const float* __restrict__ x, const float* __restrict__ g, const float* __restrict__ b,
    float* __restrict__ outF, bf16* __restrict__ outB) {
  __shared__ float red[256];
  __shared__ float red2[256];
  const int row = blockIdx.x, t = threadIdx.x;
  float v = x[(size_t)row * DMODEL + t];
  red[t] = v; red2[t] = v * v;
  __syncthreads();
  for (int s = 128; s > 0; s >>= 1) {
    if (t < s) { red[t] += red[t + s]; red2[t] += red2[t + s]; }
    __syncthreads();
  }
  float mean = red[0] * (1.0f / DMODEL);
  float var = red2[0] * (1.0f / DMODEL) - mean * mean;
  float rs = rsqrtf(var + 1e-5f);
  float y = (v - mean) * rs * g[t] + b[t];
  if (outF) outF[(size_t)row * DMODEL + t] = y;
  if (outB) outB[(size_t)row * DMODEL + t] = (bf16)y;
}

// msgin[e][d] = bf16( h[src[e]][d] + edge_attr[e][d] )
__global__ void edge_gather_kernel(const float* __restrict__ h, const float* __restrict__ ea,
                                   const int* __restrict__ src, bf16* __restrict__ out) {
  size_t tid = (size_t)blockIdx.x * blockDim.x + threadIdx.x;
  size_t e = tid >> 8;                 // D = 256
  int d = (int)(tid & 255);
  out[tid] = (bf16)(h[(size_t)src[e] * DMODEL + d] + ea[tid]);
}

// updin = bf16( (1+eps)*h + agg )
__global__ void upd_in_kernel(const float* __restrict__ h, const float* __restrict__ agg,
                              const float* __restrict__ eps, bf16* __restrict__ out) {
  size_t tid = (size_t)blockIdx.x * blockDim.x + threadIdx.x;
  out[tid] = (bf16)((1.0f + eps[0]) * h[tid] + agg[tid]);
}

// ---------------------------------------------------------------------------
// Generic bf16 WMMA GEMM: out[M,Nout] = act(A[M,K] @ W[K,Nout] + bias) (+resid)
// Block: 256 threads, 64x256 output tile. Wave w: M-tile w&3, 8 N-tiles at (w>>2)*8.
// A tile staged to LDS via GLOBAL_LOAD_ASYNC_TO_LDS_B128 (ASYNCcnt path).
// ---------------------------------------------------------------------------
#define KCHUNK 256
#define SAPAD  264
__global__ __launch_bounds__(256) void wmma_gemm_kernel(
    const bf16* __restrict__ A, const bf16* __restrict__ Wsw,
    const float* __restrict__ bias, int M, int K, int Nout, int act,
    float* __restrict__ outF, const float* __restrict__ resid,
    bf16* __restrict__ outB, int outB_trans, const int* __restrict__ scatter) {
  __shared__ bf16 sA[64][SAPAD];
  const int tid = threadIdx.x;
  const int w = tid >> 5;
  const int l = tid & 31;
  const int m0 = blockIdx.x * 64;
  const int nblk = blockIdx.y * 256;
  const int NT = Nout >> 4;
  const int mt = w & 3;
  const int nb = (w >> 2) * 8;

  v8f acc[8];
#pragma unroll
  for (int t = 0; t < 8; ++t) {
#pragma unroll
    for (int r = 0; r < 8; ++r) acc[t][r] = 0.0f;
  }

  for (int kc = 0; kc < K; kc += KCHUNK) {
    {
      int row = tid >> 2;
      int cb = (tid & 3) * 64;
      const bf16* srcp = A + (size_t)(m0 + row) * K + kc + cb;
      __builtin_prefetch(srcp + KCHUNK, 0, 1);   // global_prefetch_b8 for next K chunk
#pragma unroll
      for (int i = 0; i < 8; ++i)
        async_load_b128(srcp + i * 8, &sA[row][cb + i * 8]);   // mem -> LDS, no VGPR staging
      wait_async0();
    }
    __syncthreads();
    for (int kk = 0; kk < KCHUNK; kk += 32) {
      v16bf af = load_a_frag(&sA[mt * 16][kk], SAPAD);
      int kt = (kc + kk) >> 5;
#pragma unroll
      for (int t = 0; t < 8; ++t) {
        v16bf bfr = load_bw_frag(Wsw, NT, kt, (nblk >> 4) + nb + t);
        acc[t] = wmma_bf16(af, bfr, acc[t]);
      }
    }
    __syncthreads();
  }

  // Epilogue. C layout: VGPR r -> row r (lanes 0-15) / row 8+r (lanes 16-31), col = lane&15.
  const int half = (l >> 4) * 8;
  const int cn = l & 15;
#pragma unroll
  for (int t = 0; t < 8; ++t) {
    int col = nblk + (nb + t) * 16 + cn;
    float bv = bias ? bias[col] : 0.0f;
#pragma unroll
    for (int r = 0; r < 8; ++r) {
      int grow = m0 + mt * 16 + half + r;
      float v = acc[t][r] + bv;
      if (act == 1) v = gelu_exact(v);
      if (scatter) {
        unsafeAtomicAdd(outF + (size_t)scatter[grow] * Nout + col, v);
      } else {
        if (resid) v += resid[(size_t)grow * Nout + col];
        if (outF) outF[(size_t)grow * Nout + col] = v;
        if (outB) {
          if (outB_trans) outB[(size_t)col * M + grow] = (bf16)v;
          else            outB[(size_t)grow * Nout + col] = (bf16)v;
        }
      }
    }
  }
}

// ---------------------------------------------------------------------------
// Flash attention: one wave per (head, 16-query tile). DH=32 -> Q tile is one A fragment.
// qb,kb: [N, 256] bf16 (col = head*32+dh). vT: [256, N] bf16 (row = head*32+dh). ctx: [N,256] bf16.
// ---------------------------------------------------------------------------
__global__ __launch_bounds__(256) void attn_kernel(
    const bf16* __restrict__ qb, const bf16* __restrict__ kb,
    const bf16* __restrict__ vT, bf16* __restrict__ ctxb, int Nn) {
  __shared__ bf16 sP[8][16][40];   // per-wave 16x32 P tile, padded rows (80B, 16B aligned)
  const int w = threadIdx.x >> 5;
  const int l = threadIdx.x & 31;
  const int QT = Nn >> 4;
  const int wg = blockIdx.x * 8 + w;
  const int head = wg / QT;
  const int q0 = (wg % QT) * 16;

  const v16bf qf = load_a_frag(qb + (size_t)q0 * DMODEL + head * 32, DMODEL);

  float m[8], lsum[8];
  v8f a0, a1, vzero;
#pragma unroll
  for (int r = 0; r < 8; ++r) {
    m[r] = -1e30f; lsum[r] = 0.0f; a0[r] = 0.0f; a1[r] = 0.0f; vzero[r] = 0.0f;
  }

  const float sc = 0.17677669529663687f;  // 1/sqrt(32)
  const int half = (l >> 4) * 8;
  const int cn = l & 15;

  for (int kc = 0; kc < Nn; kc += 32) {
    v16bf bk0 = load_bt_frag(kb + (size_t)kc * DMODEL + head * 32, DMODEL);
    v16bf bk1 = load_bt_frag(kb + (size_t)(kc + 16) * DMODEL + head * 32, DMODEL);
    v8f s0 = wmma_bf16(qf, bk0, vzero);
    v8f s1 = wmma_bf16(qf, bk1, vzero);
#pragma unroll
    for (int r = 0; r < 8; ++r) {
      float x0 = s0[r] * sc, x1 = s1[r] * sc;
      float t = fmaxf(x0, x1);
      t = fmaxf(t, __shfl_xor(t, 1, 32));
      t = fmaxf(t, __shfl_xor(t, 2, 32));
      t = fmaxf(t, __shfl_xor(t, 4, 32));
      t = fmaxf(t, __shfl_xor(t, 8, 32));
      float mn = fmaxf(m[r], t);
      float al = __expf(m[r] - mn);
      float p0 = __expf(x0 - mn);
      float p1 = __expf(x1 - mn);
      float rs = p0 + p1;
      rs += __shfl_xor(rs, 1, 32);
      rs += __shfl_xor(rs, 2, 32);
      rs += __shfl_xor(rs, 4, 32);
      rs += __shfl_xor(rs, 8, 32);
      lsum[r] = lsum[r] * al + rs;
      m[r] = mn;
      a0[r] *= al; a1[r] *= al;
      sP[w][half + r][cn]      = (bf16)p0;   // keys kc..kc+15
      sP[w][half + r][16 + cn] = (bf16)p1;   // keys kc+16..kc+31
    }
    asm volatile("s_wait_dscnt 0" ::: "memory");  // LDS writes visible within wave
    v16bf pf  = load_a_frag(&sP[w][0][0], 40);
    v16bf bv0 = load_bt_frag(vT + (size_t)(head * 32 + 0)  * Nn + kc, (size_t)Nn);
    v16bf bv1 = load_bt_frag(vT + (size_t)(head * 32 + 16) * Nn + kc, (size_t)Nn);
    a0 = wmma_bf16(pf, bv0, a0);
    a1 = wmma_bf16(pf, bv1, a1);
  }

#pragma unroll
  for (int r = 0; r < 8; ++r) {
    float inv = 1.0f / lsum[r];
    size_t row = (size_t)(q0 + half + r);
    ctxb[row * DMODEL + head * 32 + cn]      = (bf16)(a0[r] * inv);
    ctxb[row * DMODEL + head * 32 + 16 + cn] = (bf16)(a1[r] * inv);
  }
}

// ---------------------------------------------------------------------------
// Host driver
// ---------------------------------------------------------------------------
extern "C" void kernel_launch(void* const* d_in, const int* in_sizes, int n_in,
                              void* d_out, int out_size, void* d_ws, size_t ws_size,
                              hipStream_t stream) {
  (void)in_sizes; (void)n_in; (void)out_size; (void)ws_size;
  const int N = NNODE, E = NEDGE, D = DMODEL, F = DFFN;

  const float* x      = (const float*)d_in[0];
  const int*   ei     = (const int*)  d_in[1];
  const int*   src    = ei;
  const int*   dst    = ei + E;
  const float* eattr  = (const float*)d_in[2];
  const float* eps    = (const float*)d_in[3];
  const float* e_w1   = (const float*)d_in[4];
  const float* e_b1   = (const float*)d_in[5];
  const float* e_w2   = (const float*)d_in[6];
  const float* e_b2   = (const float*)d_in[7];
  const float* u_w1   = (const float*)d_in[8];
  const float* u_b1   = (const float*)d_in[9];
  const float* u_w2   = (const float*)d_in[10];
  const float* u_b2   = (const float*)d_in[11];
  const float* lnl_g  = (const float*)d_in[12];
  const float* lnl_b  = (const float*)d_in[13];
  const float* lng_g  = (const float*)d_in[14];
  const float* lng_b  = (const float*)d_in[15];
  const float* lnf_g  = (const float*)d_in[16];
  const float* lnf_b  = (const float*)d_in[17];
  const float* wq     = (const float*)d_in[18];
  const float* wk     = (const float*)d_in[19];
  const float* wv     = (const float*)d_in[20];
  const float* bq     = (const float*)d_in[21];
  const float* bk     = (const float*)d_in[22];
  const float* bv     = (const float*)d_in[23];
  const float* wo     = (const float*)d_in[24];
  const float* bo     = (const float*)d_in[25];
  const float* f_w1   = (const float*)d_in[26];
  const float* f_b1   = (const float*)d_in[27];
  const float* f_w2   = (const float*)d_in[28];
  const float* f_b2   = (const float*)d_in[29];

  char* ws = (char*)d_ws;
  size_t off = 0;
  auto alloc = [&](size_t bytes) -> char* {
    char* p = ws + off;
    off += (bytes + 255) & ~(size_t)255;
    return p;
  };

  bf16* sw_e1 = (bf16*)alloc((size_t)D * D * 2);
  bf16* sw_e2 = (bf16*)alloc((size_t)D * D * 2);
  bf16* sw_u1 = (bf16*)alloc((size_t)D * D * 2);
  bf16* sw_u2 = (bf16*)alloc((size_t)D * D * 2);
  bf16* sw_q  = (bf16*)alloc((size_t)D * D * 2);
  bf16* sw_k  = (bf16*)alloc((size_t)D * D * 2);
  bf16* sw_v  = (bf16*)alloc((size_t)D * D * 2);
  bf16* sw_o  = (bf16*)alloc((size_t)D * D * 2);
  bf16* sw_f1 = (bf16*)alloc((size_t)D * F * 2);
  bf16* sw_f2 = (bf16*)alloc((size_t)F * D * 2);

  float* h    = (float*)alloc((size_t)N * D * 4);
  float* agg  = (float*)alloc((size_t)N * D * 4);
  float* x1   = (float*)alloc((size_t)N * D * 4);
  float* x2   = (float*)alloc((size_t)N * D * 4);

  bf16* msgin  = (bf16*)alloc((size_t)E * D * 2);   // 64 MB
  bf16* msgmid = (bf16*)alloc((size_t)E * D * 2);   // 64 MB

  // Reuse the msgin region for all post-aggregation bf16 buffers (2 MB each slot).
  char* rb = (char*)msgin;
  const size_t SL = (size_t)N * D * 2;
  bf16* updin  = (bf16*)(rb + 0 * SL);
  bf16* updmid = (bf16*)(rb + 1 * SL);
  bf16* gbf    = (bf16*)(rb + 2 * SL);
  bf16* qb     = (bf16*)(rb + 3 * SL);
  bf16* kb     = (bf16*)(rb + 4 * SL);
  bf16* vT     = (bf16*)(rb + 5 * SL);
  bf16* ctxb   = (bf16*)(rb + 6 * SL);
  bf16* ffnin  = (bf16*)(rb + 7 * SL);
  bf16* ffnmid = (bf16*)(rb + 8 * SL);              // needs 4 slots (N*F*2 = 8 MB)

  auto swz = [&](const float* W, bf16* o, int K, int Nout) {
    size_t tot = (size_t)K * Nout;
    swizzle_w_kernel<<<dim3((unsigned)(tot / 256)), dim3(256), 0, stream>>>(W, o, K, Nout);
  };
  auto gemm = [&](const bf16* A, const bf16* Wsw, const float* bias, int M, int K, int Nout,
                  int act, float* outF, const float* resid, bf16* outB, int outTrans,
                  const int* scat) {
    dim3 g((unsigned)(M / 64), (unsigned)(Nout / 256));
    wmma_gemm_kernel<<<g, 256, 0, stream>>>(A, Wsw, bias, M, K, Nout, act,
                                            outF, resid, outB, outTrans, scat);
  };

  // 0) weight swizzle f32 -> bf16 fragment order
  swz(e_w1, sw_e1, D, D); swz(e_w2, sw_e2, D, D);
  swz(u_w1, sw_u1, D, D); swz(u_w2, sw_u2, D, D);
  swz(wq, sw_q, D, D); swz(wk, sw_k, D, D); swz(wv, sw_v, D, D); swz(wo, sw_o, D, D);
  swz(f_w1, sw_f1, D, F); swz(f_w2, sw_f2, F, D);

  // 1) h = LN_local(x)
  layernorm_kernel<<<N, 256, 0, stream>>>(x, lnl_g, lnl_b, h, nullptr);

  // 2) zero aggregation buffer
  hipMemsetAsync(agg, 0, (size_t)N * D * 4, stream);

  // 3) edge messages: msgin = bf16(h[src] + edge_attr)
  edge_gather_kernel<<<dim3((unsigned)((size_t)E * D / 256)), 256, 0, stream>>>(h, eattr, src, msgin);

  // 4) edge MLP, fused scatter-add into agg[dst]
  gemm(msgin, sw_e1, e_b1, E, D, D, 1, nullptr, nullptr, msgmid, 0, nullptr);
  gemm(msgmid, sw_e2, e_b2, E, D, D, 0, agg, nullptr, nullptr, 0, dst);

  // 5) update MLP: x1 = x + mlp2((1+eps)*h + agg)
  upd_in_kernel<<<dim3((unsigned)((size_t)N * D / 256)), 256, 0, stream>>>(h, agg, eps, updin);
  gemm(updin, sw_u1, u_b1, N, D, D, 1, nullptr, nullptr, updmid, 0, nullptr);
  gemm(updmid, sw_u2, u_b2, N, D, D, 0, x1, x, nullptr, 0, nullptr);

  // 6) global attention
  layernorm_kernel<<<N, 256, 0, stream>>>(x1, lng_g, lng_b, nullptr, gbf);
  gemm(gbf, sw_q, bq, N, D, D, 0, nullptr, nullptr, qb, 0, nullptr);
  gemm(gbf, sw_k, bk, N, D, D, 0, nullptr, nullptr, kb, 0, nullptr);
  gemm(gbf, sw_v, bv, N, D, D, 0, nullptr, nullptr, vT, 1, nullptr);  // transposed store [D,N]
  attn_kernel<<<dim3((unsigned)(NHEAD * (N / 16) / 8)), 256, 0, stream>>>(qb, kb, vT, ctxb, N);
  gemm(ctxb, sw_o, bo, N, D, D, 0, x2, x1, nullptr, 0, nullptr);

  // 7) FFN: out = x2 + mlp2(LN_ffn(x2))
  layernorm_kernel<<<N, 256, 0, stream>>>(x2, lnf_g, lnf_b, nullptr, ffnin);
  gemm(ffnin, sw_f1, f_b1, N, D, F, 1, nullptr, nullptr, ffnmid, 0, nullptr);
  gemm(ffnmid, sw_f2, f_b2, N, F, D, 0, (float*)d_out, x2, nullptr, 0, nullptr);
}
